// LoRAExpert_54047868453052
// MI455X (gfx1250) — matrix operand order repaired and checked by reading, not compiled, not run on hardware
//
#include <hip/hip_runtime.h>
#include <hip/hip_bf16.h>
#include <math.h>

// ---------------------------------------------------------------------------
// ViT-Base forward with expert LoRA on K/V (first 5 blocks), CDNA5 (gfx1250).
// All GEMMs run on v_wmma_f32_16x16x32_f16 (wave32), fp32 accumulate.
// 64x64 block tile, 8 waves, each wave computes 16x32 (2 WMMAs / K-step,
// A-fragment reuse). Staging is branch-free: clamped-address float4 loads +
// cndmask zeroing, packed 8xf16 LDS stores.
// ---------------------------------------------------------------------------

typedef __attribute__((ext_vector_type(16))) _Float16 v16h;
typedef __attribute__((ext_vector_type(8)))  _Float16 v8h;
typedef __attribute__((ext_vector_type(8)))  float    v8f;

#define BM 64
#define BN 64
#define BK 32
#define LDP 8    // LDS row padding (halves); row stride 40 halves = 80B (16B-aligned)

__device__ __forceinline__ int imin(int a, int b) { return a < b ? a : b; }

// Stage a 64-row x 32-col fp32 tile (contiguous in k) into LDS as f16.
// Branch-free: rows clamped (padding rows harmless), k-overhang zeroed by select.
__device__ __forceinline__ void stage_rowmajor(
    _Float16 (*dst)[BK + LDP], const float* __restrict__ src,
    long long rowStride, int row0, int maxRow, int k0, int K)
{
  const int tid = threadIdx.x;          // 256 threads; 8 floats each
  const int r   = tid >> 2;             // 0..63
  const int kq  = (tid & 3) * 8;        // 0,8,16,24
  const int gr  = imin(row0 + r, maxRow - 1);
  const float* p = src + (long long)gr * rowStride + (k0 + kq);
  float4 a0 = *reinterpret_cast<const float4*>(p);
  float4 a1 = *reinterpret_cast<const float4*>(p + 4);
  float vals[8] = {a0.x, a0.y, a0.z, a0.w, a1.x, a1.y, a1.z, a1.w};
  v8h h;
#pragma unroll
  for (int j = 0; j < 8; ++j) {
    float f = (k0 + kq + j < K) ? vals[j] : 0.f;
    h[j] = (_Float16)f;
  }
  *reinterpret_cast<v8h*>(&dst[r][kq]) = h;
}

// Generic tiled GEMM core:  out[m,n] (+)= act( alpha * sum_k A[m,k]*W[n,k] + bias[n] )
// W addressed as W[n*wRS + k*wCS]; wCS==1 takes the vectorized path.
__device__ __forceinline__ void gemm_core(
    const float* __restrict__ A, int lda,
    const float* __restrict__ W, long long wRS, long long wCS,
    const float* __restrict__ bias,
    float* __restrict__ Cout, int ldc,
    int M, int N, int K, float alpha, int accumulate, int act)
{
  __shared__ _Float16 Ah[BM][BK + LDP];
  __shared__ _Float16 Wh[BN][BK + LDP];

  const int tid   = threadIdx.x;        // 0..255 (8 waves)
  const int lane  = tid & 31;
  const int wid   = tid >> 5;
  const int wm    = wid & 3;            // wave M sub-tile (0..3)
  const int wn    = wid >> 2;           // wave N half (0..1), covers 32 cols
  const int tileM = blockIdx.y * BM;
  const int tileN = blockIdx.x * BN;

  v8f acc0 = {0.f,0.f,0.f,0.f,0.f,0.f,0.f,0.f};
  v8f acc1 = {0.f,0.f,0.f,0.f,0.f,0.f,0.f,0.f};

  for (int k0 = 0; k0 < K; k0 += BK) {
    stage_rowmajor(Ah, A, lda, tileM, M, k0, K);
    if (wCS == 1) {
      stage_rowmajor(Wh, W, wRS, tileN, N, k0, K);
    } else {
      // gather path (P.V): 8 independent scalar loads, branch-free
      const int r  = tid >> 2;                       // 0..63 (output col n)
      const int kq = (tid & 3) * 8;
      const int gn = imin(tileN + r, N - 1);
#pragma unroll
      for (int j = 0; j < 8; ++j) {
        int gk = imin(k0 + kq + j, K - 1);
        float f = W[(long long)gn * wRS + (long long)gk * wCS];
        f = (k0 + kq + j < K) ? f : 0.f;
        Wh[r][kq + j] = (_Float16)f;
      }
    }
    // prefetch next A K-tile (global_prefetch_b8)
    if (k0 + BK < K) {
      int pm = imin(tileM + (tid & 63), M - 1);
      __builtin_prefetch(&A[(long long)pm * lda + k0 + BK], 0, 1);
    }
    __syncthreads();

    // ---- fragments per wave32 16-bit WMMA layouts ----
    v16h afrag, bfrag0, bfrag1;
    const int mrow  = wm * 16 + (lane & 15);
    const int ncol0 = wn * 32 + (lane & 15);
    const int ncol1 = ncol0 + 16;
    const int hv    = lane >> 4;
#pragma unroll
    for (int vi = 0; vi < 8; ++vi) {
      // A 16x32: VGPR 0-3 -> K = hv*8 + 2vi ; VGPR 4-7 -> K = 16 + hv*8 + 2(vi-4)
      int ka = (vi < 4) ? (hv * 8 + 2 * vi) : (16 + hv * 8 + 2 * (vi - 4));
      afrag[2 * vi]     = Ah[mrow][ka];
      afrag[2 * vi + 1] = Ah[mrow][ka + 1];
      // B 32x16: same K order as A
      int kb = ka;
      bfrag0[2 * vi]     = Wh[ncol0][kb];
      bfrag0[2 * vi + 1] = Wh[ncol0][kb + 1];
      bfrag1[2 * vi]     = Wh[ncol1][kb];
      bfrag1[2 * vi + 1] = Wh[ncol1][kb + 1];
    }

    acc0 = __builtin_amdgcn_wmma_f32_16x16x32_f16(false, afrag, false, bfrag0,
                                                  (short)0, acc0, false, false);
    acc1 = __builtin_amdgcn_wmma_f32_16x16x32_f16(false, afrag, false, bfrag1,
                                                  (short)0, acc1, false, false);
    __syncthreads();
  }

  // ---- epilogue: C/D layout -> lane n = L%16, VGPR vi -> m = vi + 8*(L/16) ----
  const int hv = lane >> 4;
#pragma unroll
  for (int t = 0; t < 2; ++t) {
    const v8f& acc = t ? acc1 : acc0;
    int gn = tileN + wn * 32 + t * 16 + (lane & 15);
    if (gn < N) {
      const float bv = bias ? bias[gn] : 0.f;
#pragma unroll
      for (int vi = 0; vi < 8; ++vi) {
        int gm = tileM + wm * 16 + vi + 8 * hv;
        if (gm < M) {
          float r = acc[vi] * alpha + bv;
          if (act == 1) r = 0.5f * r * (1.f + erff(r * 0.70710678118f)); // exact GELU
          long long idx = (long long)gm * ldc + gn;
          if (accumulate) Cout[idx] += r; else Cout[idx] = r;
        }
      }
    }
  }
}

// ---- thin wrappers -----------------------------------------------------------

__global__ void gemm_kernel(const float* A, int lda, const float* W,
                            long long wRS, long long wCS, const float* bias,
                            float* out, int ldc, int M, int N, int K,
                            float alpha, int accum, int act) {
  gemm_core(A, lda, W, wRS, wCS, bias, out, ldc, M, N, K, alpha, accum, act);
}

// LoRA stage 1: t[b] = xn[b] @ A_e^T   (per-sample expert select)
__global__ void lora1_kernel(const float* xn, const float* lA, const int* eids,
                             float* t, int layer) {
  int b = blockIdx.z;
  int e = eids[b];
  const float* Ap = xn + (long long)b * 197 * 768;
  const float* Wp = lA + ((long long)layer * 4 + e) * 64 * 768;
  float* out = t + (long long)b * 197 * 64;
  gemm_core(Ap, 768, Wp, 768, 1, nullptr, out, 64, 197, 64, 768, 1.f, 0, 0);
}

// LoRA stage 2: qkv[b][:, colOff:colOff+768] += (t[b] @ B_e^T) * (alpha/R)
__global__ void lora2_kernel(const float* t, const float* lB, const int* eids,
                             float* qkv, int layer, int colOff) {
  int b = blockIdx.z;
  int e = eids[b];
  const float* Ap = t + (long long)b * 197 * 64;
  const float* Wp = lB + ((long long)layer * 4 + e) * 768 * 64;
  float* out = qkv + (long long)b * 197 * 2304 + colOff;
  gemm_core(Ap, 64, Wp, 64, 1, nullptr, out, 2304, 197, 768, 64, 1.f / 64.f, 1, 0);
}

// Attention scores: S[b,h] = (Q_bh @ K_bh^T) * 1/8  (strided views into qkv)
__global__ void attn_scores_kernel(const float* qkv, float* scores) {
  int z = blockIdx.z, b = z / 12, h = z % 12;
  const float* Q  = qkv + (long long)b * 197 * 2304 + h * 64;          // lda 2304
  const float* Kp = qkv + (long long)b * 197 * 2304 + 768 + h * 64;    // row-stride 2304
  float* out = scores + (long long)z * 197 * 208;
  gemm_core(Q, 2304, Kp, 2304, 1, nullptr, out, 208, 197, 197, 64, 0.125f, 0, 0);
}

// O[b,h] = P @ V  (V accessed transposed: W[d, n] = V[n, d])
__global__ void attn_pv_kernel(const float* scores, const float* qkv, float* attno) {
  int z = blockIdx.z, b = z / 12, h = z % 12;
  const float* P = scores + (long long)z * 197 * 208;
  const float* V = qkv + (long long)b * 197 * 2304 + 1536 + h * 64;    // wRS=1, wCS=2304
  float* out = attno + (long long)b * 197 * 768 + h * 64;
  gemm_core(P, 208, V, 1, 2304, nullptr, out, 768, 197, 64, 197, 1.f, 0, 0);
}

// ---- pointwise / reduction kernels ------------------------------------------

__device__ __forceinline__ void ln_row(const float* xr, const float* g,
                                       const float* b, float* outr) {
  __shared__ float sb[256];
  int tid = threadIdx.x;
  float s = 0.f;
  for (int c = tid; c < 768; c += 256) s += xr[c];
  sb[tid] = s; __syncthreads();
  for (int st = 128; st > 0; st >>= 1) { if (tid < st) sb[tid] += sb[tid + st]; __syncthreads(); }
  float mean = sb[0] * (1.f / 768.f); __syncthreads();
  float vs = 0.f;
  for (int c = tid; c < 768; c += 256) { float d = xr[c] - mean; vs += d * d; }
  sb[tid] = vs; __syncthreads();
  for (int st = 128; st > 0; st >>= 1) { if (tid < st) sb[tid] += sb[tid + st]; __syncthreads(); }
  float inv = rsqrtf(sb[0] * (1.f / 768.f) + 1e-6f); __syncthreads();
  for (int c = tid; c < 768; c += 256) outr[c] = (xr[c] - mean) * inv * g[c] + b[c];
}

__global__ void layernorm_kernel(const float* x, const float* g, const float* b,
                                 float* out) {
  long long row = blockIdx.x;
  ln_row(x + row * 768, g, b, out + row * 768);
}

__global__ void final_ln_kernel(const float* x, const float* g, const float* b,
                                float* out) {
  int bi = blockIdx.x;                        // CLS token: row (b, 0)
  ln_row(x + (long long)bi * 197 * 768, g, b, out + (long long)bi * 768);
}

__global__ void softmax_kernel(float* scores) {
  int z = blockIdx.x / 197, m = blockIdx.x % 197;
  float* p = scores + (long long)z * 197 * 208 + (long long)m * 208;
  __shared__ float sb[256];
  int tid = threadIdx.x;
  float v = (tid < 197) ? p[tid] : -3.4e38f;
  sb[tid] = v; __syncthreads();
  for (int st = 128; st > 0; st >>= 1) { if (tid < st) sb[tid] = fmaxf(sb[tid], sb[tid + st]); __syncthreads(); }
  float mx = sb[0]; __syncthreads();
  float e = (tid < 197) ? expf(v - mx) : 0.f;
  sb[tid] = e; __syncthreads();
  for (int st = 128; st > 0; st >>= 1) { if (tid < st) sb[tid] += sb[tid + st]; __syncthreads(); }
  float inv = 1.f / sb[0];
  if (tid < 197) p[tid] = e * inv;
}

// Unfold 224x224x3 images into [B*196, 768] patch rows (c*256 + py*16 + px)
__global__ void unfold_kernel(const float* in, float* patches) {
  int row = blockIdx.x;                       // b*196 + gy*14 + gx
  int b = row / 196, pidx = row % 196, gy = pidx / 14, gx = pidx % 14;
  for (int col = threadIdx.x; col < 768; col += 256) {
    int c = col >> 8, py = (col >> 4) & 15, px = col & 15;
    patches[(long long)row * 768 + col] =
        in[(((long long)b * 3 + c) * 224 + gy * 16 + py) * 224 + gx * 16 + px];
  }
}

// x[b,0] = cls + pos[0];  x[b,1+p] = patch_embed[b,p] + pos[1+p]
__global__ void assemble_x_kernel(const float* pe, const float* cls,
                                  const float* pos, float* x) {
  int row = blockIdx.x;                       // b*197 + n
  int b = row / 197, n = row % 197;
  for (int c = threadIdx.x; c < 768; c += 256) {
    float base = (n == 0) ? cls[c] : pe[((long long)b * 196 + (n - 1)) * 768 + c];
    x[(long long)row * 768 + c] = base + pos[(long long)n * 768 + c];
  }
}

// ---------------------------------------------------------------------------

extern "C" void kernel_launch(void* const* d_in, const int* in_sizes, int n_in,
                              void* d_out, int out_size, void* d_ws, size_t ws_size,
                              hipStream_t stream) {
  const float* inputs  = (const float*)d_in[0];
  const int*   eids    = (const int*)d_in[1];
  const float* patch_w = (const float*)d_in[2];
  const float* patch_b = (const float*)d_in[3];
  const float* clsTok  = (const float*)d_in[4];
  const float* pos     = (const float*)d_in[5];
  auto bp = [&](int i, int j) { return (const float*)d_in[6 + i * 12 + j]; };
  const float* norm_g  = (const float*)d_in[150];
  const float* norm_b  = (const float*)d_in[151];
  const float* lkA     = (const float*)d_in[152];
  const float* lkB     = (const float*)d_in[153];
  const float* lvA     = (const float*)d_in[154];
  const float* lvB     = (const float*)d_in[155];
  float* outp = (float*)d_out;

  // workspace layout (floats); each region gets a 256-float tail pad so
  // clamped-row / K-overhang staging reads stay inside the workspace.
  const long long PADF = 256;
  float* ws = (float*)d_ws;
  long long off = 0;
  float* x      = ws + off; off += 3152LL * 768  + PADF;   // residual stream
  float* xn     = ws + off; off += 3152LL * 768  + PADF;   // layernorm output
  float* qkvb   = ws + off; off += 3152LL * 2304 + PADF;   // fused qkv
  float* attno  = ws + off; off += 3152LL * 768  + PADF;   // attention output
  float* hbuf   = ws + off; off += 3152LL * 3072 + PADF;   // MLP hidden
  float* tlora  = ws + off; off += 3152LL * 64   + PADF;   // LoRA rank-64 temp
  float* scores = ws + off; off += 192LL * 197 * 208 + PADF;
  float* patches = hbuf;    // pre-loop aliases (regions free until block 0)
  float* pe      = scores;

  dim3 blk(256);
  auto gg = [](int M, int N, int Z) { return dim3((N + BN - 1) / BN, (M + BM - 1) / BM, Z); };

  // patch embedding
  unfold_kernel<<<16 * 196, blk, 0, stream>>>(inputs, patches);
  gemm_kernel<<<gg(3136, 768, 1), blk, 0, stream>>>(
      patches, 768, patch_w, 768, 1, patch_b, pe, 768, 3136, 768, 768, 1.f, 0, 0);
  assemble_x_kernel<<<16 * 197, blk, 0, stream>>>(pe, clsTok, pos, x);

  for (int i = 0; i < 12; ++i) {
    // ---- attention ----
    layernorm_kernel<<<3152, blk, 0, stream>>>(x, bp(i, 0), bp(i, 1), xn);
    gemm_kernel<<<gg(3152, 2304, 1), blk, 0, stream>>>(
        xn, 768, bp(i, 2), 768, 1, bp(i, 3), qkvb, 2304, 3152, 2304, 768, 1.f, 0, 0);

    if (i < 5) {  // expert LoRA on K and V
      lora1_kernel<<<gg(197, 64, 16), blk, 0, stream>>>(xn, lkA, eids, tlora, i);
      lora2_kernel<<<gg(197, 768, 16), blk, 0, stream>>>(tlora, lkB, eids, qkvb, i, 768);
      lora1_kernel<<<gg(197, 64, 16), blk, 0, stream>>>(xn, lvA, eids, tlora, i);
      lora2_kernel<<<gg(197, 768, 16), blk, 0, stream>>>(tlora, lvB, eids, qkvb, i, 1536);
    }

    attn_scores_kernel<<<gg(197, 197, 192), blk, 0, stream>>>(qkvb, scores);
    softmax_kernel<<<192 * 197, blk, 0, stream>>>(scores);
    attn_pv_kernel<<<gg(197, 64, 192), blk, 0, stream>>>(scores, qkvb, attno);

    gemm_kernel<<<gg(3152, 768, 1), blk, 0, stream>>>(           // x += proj
        attno, 768, bp(i, 4), 768, 1, bp(i, 5), x, 768, 3152, 768, 768, 1.f, 1, 0);

    // ---- MLP ----
    layernorm_kernel<<<3152, blk, 0, stream>>>(x, bp(i, 6), bp(i, 7), xn);
    gemm_kernel<<<gg(3152, 3072, 1), blk, 0, stream>>>(          // GELU(fc1)
        xn, 768, bp(i, 8), 768, 1, bp(i, 9), hbuf, 3072, 3152, 3072, 768, 1.f, 0, 1);
    gemm_kernel<<<gg(3152, 768, 1), blk, 0, stream>>>(           // x += fc2
        hbuf, 3072, bp(i, 10), 3072, 1, bp(i, 11), x, 768, 3152, 768, 3072, 1.f, 1, 0);
  }

  final_ln_kernel<<<16, blk, 0, stream>>>(x, norm_g, norm_b, outp);
}
